// HRformer_86595130622491
// MI455X (gfx1250) — matrix operand order repaired and checked by reading, not compiled, not run on hardware
//
#include <hip/hip_runtime.h>
#include <math.h>

// ---------------------------------------------------------------------------
// Types for CDNA5 WMMA (wave32): v_wmma_f32_16x16x32_bf16
// ---------------------------------------------------------------------------
typedef __bf16 bf16_t;
typedef bf16_t bf16x16 __attribute__((ext_vector_type(16)));
typedef float  f32x8   __attribute__((ext_vector_type(8)));

union Frag { bf16x16 v; unsigned int u[8]; };

__device__ __forceinline__ unsigned int pack2(float a, float b) {
  union { unsigned int u; bf16_t h[2]; } x;
  x.h[0] = (bf16_t)a; x.h[1] = (bf16_t)b;
  return x.u;
}

__device__ __forceinline__ float actf(float v, int act) {
  if (act == 1) return fmaxf(v, 0.f);
  if (act == 2) return 0.5f * v * (1.f + erff(v * 0.70710678118f));  // exact gelu
  if (act == 3) return 1.f / (1.f + __expf(-v));
  return v;
}

// ---------------------------------------------------------------------------
// Generic batched GEMM: C = act(A @ B(+T) + bias [+ C])   (bf16 WMMA, f32 acc)
// Block = 256 threads = 8 waves, 4(M) x 2(N) wave grid -> 128x64 block tile.
// Each wave: 2 A-frags x 2 B-frags = 4 v_wmma per 32-wide K step.
// LDS holds packed bf16 pairs (one uint = 2 consecutive K values) so fragment
// construction is a direct 32-bit LDS read per VGPR — no repacking ALU.
// ---------------------------------------------------------------------------
struct GemmP {
  const float* A; const float* B; const float* bias; float* C;
  long long sAo, sAi, sBo, sBi, sCo, sCi;   // outer/inner batch strides
  int M, N, K, lda, ldb, ldc;
  int inner, opBT, act, beta;
};

__global__ __launch_bounds__(256) void gemm_k(GemmP p) {
  __shared__ unsigned int AsU[128 * 16];   // [m][kpair]
  __shared__ unsigned int BsU[64 * 16];    // [n][kpair]
  const int bz = blockIdx.z;
  const int bo = bz / p.inner, bi = bz % p.inner;
  const float* A = p.A + (long long)bo * p.sAo + (long long)bi * p.sAi;
  const float* B = p.B + (long long)bo * p.sBo + (long long)bi * p.sBi;
  float*       C = p.C + (long long)bo * p.sCo + (long long)bi * p.sCi;
  const int tileM = blockIdx.y * 128, tileN = blockIdx.x * 64;
  const int tid = threadIdx.x;
  const int wave = tid >> 5, lane = tid & 31;
  const int wm = wave >> 1, wn = wave & 1;
  const int mloc = lane & 15, khalf = lane >> 4;

  f32x8 acc00 = {0.f,0.f,0.f,0.f,0.f,0.f,0.f,0.f};
  f32x8 acc01 = acc00, acc10 = acc00, acc11 = acc00;

  const bool fullM = (tileM + 128 <= p.M);
  const bool fullN = (tileN + 64 <= p.N);

  for (int k0 = 0; k0 < p.K; k0 += 32) {
    const bool fullK = (k0 + 32 <= p.K);

    // ---- stage A: 128 x 32 fp32 -> 128 x 16 bf16-pairs ----
    if (fullM && fullK) {
#pragma unroll
      for (int j = 0; j < 8; ++j) {
        int e = tid + j * 256;
        int r = e >> 4, kp = e & 15;
        const float* src = A + (long long)(tileM + r) * p.lda + k0 + 2 * kp;
        AsU[e] = pack2(src[0], src[1]);
      }
    } else {
#pragma unroll
      for (int j = 0; j < 8; ++j) {
        int e = tid + j * 256;
        int r = e >> 4, kp = e & 15;
        int gr = tileM + r, k = k0 + 2 * kp;
        float f0 = 0.f, f1 = 0.f;
        if (gr < p.M) {
          const float* src = A + (long long)gr * p.lda;
          if (k < p.K)     f0 = src[k];
          if (k + 1 < p.K) f1 = src[k + 1];
        }
        AsU[e] = pack2(f0, f1);
      }
    }

    // ---- stage B: 32 x 64 fp32 -> [n][kpair] bf16-pairs ----
    if (fullN && fullK) {
      if (p.opBT) {
#pragma unroll
        for (int j = 0; j < 4; ++j) {
          int e = tid + j * 256;
          int n = e >> 4, kp = e & 15;
          const float* src = B + (long long)(tileN + n) * p.ldb + k0 + 2 * kp;
          BsU[e] = pack2(src[0], src[1]);
        }
      } else {
#pragma unroll
        for (int j = 0; j < 4; ++j) {
          int e = tid + j * 256;
          int n = e >> 4, kp = e & 15;
          const float* src = B + (long long)(k0 + 2 * kp) * p.ldb + tileN + n;
          BsU[e] = pack2(src[0], src[p.ldb]);
        }
      }
    } else {
#pragma unroll
      for (int j = 0; j < 4; ++j) {
        int e = tid + j * 256;
        int n = e >> 4, kp = e & 15;
        int gn = tileN + n, k = k0 + 2 * kp;
        float f0 = 0.f, f1 = 0.f;
        if (gn < p.N) {
          if (p.opBT) {
            const float* src = B + (long long)gn * p.ldb;
            if (k < p.K)     f0 = src[k];
            if (k + 1 < p.K) f1 = src[k + 1];
          } else {
            if (k < p.K)     f0 = B[(long long)k * p.ldb + gn];
            if (k + 1 < p.K) f1 = B[(long long)(k + 1) * p.ldb + gn];
          }
        }
        BsU[e] = pack2(f0, f1);
      }
    }

    if (k0 + 32 < p.K) {                        // hint next K slab into cache
      int gr = tileM + (tid >> 1);
      if (gr < p.M) __builtin_prefetch(&A[(long long)gr * p.lda + k0 + 32], 0, 0);
    }
    __syncthreads();

    Frag a0, a1, b0, b1;
#pragma unroll
    for (int vi = 0; vi < 8; ++vi) {
      // A-frag kpair index (ISA 7.12.2, 16-bit A 16x32 layout)
      int kpa = (vi < 4) ? (khalf * 4 + vi) : (8 + khalf * 4 + (vi - 4));
      a0.u[vi] = AsU[(wm * 32 + mloc) * 16 + kpa];
      a1.u[vi] = AsU[(wm * 32 + 16 + mloc) * 16 + kpa];
      // B-frag kpair index (16-bit B 32x16 layout)
      int kpb = kpa;
      b0.u[vi] = BsU[(wn * 32 + mloc) * 16 + kpb];
      b1.u[vi] = BsU[(wn * 32 + 16 + mloc) * 16 + kpb];
    }
    acc00 = __builtin_amdgcn_wmma_f32_16x16x32_bf16(false, a0.v, false, b0.v, (short)0, acc00, false, false);
    acc01 = __builtin_amdgcn_wmma_f32_16x16x32_bf16(false, a0.v, false, b1.v, (short)0, acc01, false, false);
    acc10 = __builtin_amdgcn_wmma_f32_16x16x32_bf16(false, a1.v, false, b0.v, (short)0, acc10, false, false);
    acc11 = __builtin_amdgcn_wmma_f32_16x16x32_bf16(false, a1.v, false, b1.v, (short)0, acc11, false, false);
    __syncthreads();
  }

  auto emit = [&](const f32x8& acc, int rb, int cb) {
#pragma unroll
    for (int r = 0; r < 8; ++r) {            // C layout: VGPR r, halves = +8 rows
      int row = rb + r + 8 * khalf;
      int col = cb + mloc;
      if (row < p.M && col < p.N) {
        float v = acc[r];
        if (p.bias) v += p.bias[col];
        long long idx = (long long)row * p.ldc + col;
        if (p.beta) v += C[idx];
        C[idx] = actf(v, p.act);
      }
    }
  };
  emit(acc00, tileM + wm * 32,      tileN + wn * 32);
  emit(acc01, tileM + wm * 32,      tileN + wn * 32 + 16);
  emit(acc10, tileM + wm * 32 + 16, tileN + wn * 32);
  emit(acc11, tileM + wm * 32 + 16, tileN + wn * 32 + 16);
}

// ---------------------------------------------------------------------------
// Flash-style MHA: one wave per 16-row q-tile, Dh=16 (zero-padded to K=32),
// online softmax via width-16 shuffles matching the WMMA C layout.
// ---------------------------------------------------------------------------
struct FlashP {
  const float* Q; const float* K; const float* V; float* O;
  long long sQn, sKn, sVn, sOn;
  int qrs, krs, vrs, ors;
  int N, T, H;
};

__global__ __launch_bounds__(256) void flash_k(FlashP p) {
  __shared__ bf16_t Pt[8][256];
  const int wave = threadIdx.x >> 5, lane = threadIdx.x & 31;
  const int mloc = lane & 15, khalf = lane >> 4;
  const int nQt = (p.T + 15) >> 4;
  const long long total = (long long)p.N * p.H * nQt;
  long long w = (long long)blockIdx.x * 8 + wave;
  const bool active = (w < total);
  if (!active) w = 0;
  const int qt = (int)(w % nQt);
  const int h  = (int)((w / nQt) % p.H);
  const int n  = (int)(w / ((long long)nQt * p.H));
  const float* Qp = p.Q + (long long)n * p.sQn + h * 16;
  const float* Kp = p.K + (long long)n * p.sKn + h * 16;
  const float* Vp = p.V + (long long)n * p.sVn + h * 16;
  const int q0 = qt * 16;
  const float scale = 0.25f;                 // 1/sqrt(Dh=16), folded into Q

  Frag aq;
#pragma unroll
  for (int vi = 0; vi < 8; ++vi) {
    int kk = (vi < 4) ? (khalf * 8 + 2 * vi) : (16 + khalf * 8 + 2 * (vi - 4));
    float f0 = 0.f, f1 = 0.f;
    if (kk < 16) {
      int qr = q0 + mloc; if (qr >= p.T) qr = p.T - 1;
      const float* qrow = Qp + (long long)qr * p.qrs;
      f0 = qrow[kk] * scale; f1 = qrow[kk + 1] * scale;
    }
    aq.u[vi] = pack2(f0, f1);
  }

  float mrow[8], lrow[8];
#pragma unroll
  for (int r = 0; r < 8; ++r) { mrow[r] = -3.0e38f; lrow[r] = 0.f; }
  f32x8 oacc = {0.f,0.f,0.f,0.f,0.f,0.f,0.f,0.f};

  const int nKt = (p.T + 15) >> 4;
  for (int kt = 0; kt < nKt; ++kt) {
    Frag bk;
#pragma unroll
    for (int vi = 0; vi < 8; ++vi) {         // B = K^T : [dh][key]
      int kk = (vi < 4) ? (khalf * 8 + 2 * vi) : (16 + khalf * 8 + 2 * (vi - 4));
      float f0 = 0.f, f1 = 0.f;
      if (kk < 16) {
        int key = kt * 16 + mloc;
        if (key < p.T) {
          const float* krow = Kp + (long long)key * p.krs;
          f0 = krow[kk]; f1 = krow[kk + 1];
        }
      }
      bk.u[vi] = pack2(f0, f1);
    }
    f32x8 z8 = {0.f,0.f,0.f,0.f,0.f,0.f,0.f,0.f};
    f32x8 s = __builtin_amdgcn_wmma_f32_16x16x32_bf16(false, aq.v, false, bk.v, (short)0, z8, false, false);

    const bool colvalid = (kt * 16 + mloc) < p.T;
#pragma unroll
    for (int r = 0; r < 8; ++r) {
      float sv = colvalid ? s[r] : -3.0e38f;
      float mt = sv;
      mt = fmaxf(mt, __shfl_xor(mt, 1, 16));
      mt = fmaxf(mt, __shfl_xor(mt, 2, 16));
      mt = fmaxf(mt, __shfl_xor(mt, 4, 16));
      mt = fmaxf(mt, __shfl_xor(mt, 8, 16));
      float mn = fmaxf(mrow[r], mt);
      float pr = __expf(sv - mn);
      float al = __expf(mrow[r] - mn);
      mrow[r] = mn;
      float ps = pr;
      ps += __shfl_xor(ps, 1, 16);
      ps += __shfl_xor(ps, 2, 16);
      ps += __shfl_xor(ps, 4, 16);
      ps += __shfl_xor(ps, 8, 16);
      lrow[r] = lrow[r] * al + ps;
      oacc[r] *= al;
      s[r] = pr;
    }
    __syncthreads();
#pragma unroll
    for (int r = 0; r < 8; ++r)              // C layout -> LDS row-major P
      Pt[wave][(r + 8 * khalf) * 16 + mloc] = (bf16_t)s[r];
    __syncthreads();

    Frag ap, bv;
#pragma unroll
    for (int vi = 0; vi < 8; ++vi) {         // P back as A-frag (K<=16)
      if (vi < 4) {
        int kk = khalf * 8 + 2 * vi;
        ap.u[vi] = *(const unsigned int*)&Pt[wave][mloc * 16 + kk];
      } else ap.u[vi] = 0u;
    }
#pragma unroll
    for (int vi = 0; vi < 8; ++vi) {         // B = V : [key][dh]
      int kk = (vi < 4) ? (khalf * 8 + 2 * vi) : (16 + khalf * 8 + 2 * (vi - 4));
      float f0 = 0.f, f1 = 0.f;
      if (kk < 16) {
        int k0_ = kt * 16 + kk;
        if (k0_ < p.T)     f0 = Vp[(long long)k0_ * p.vrs + mloc];
        if (k0_ + 1 < p.T) f1 = Vp[(long long)(k0_ + 1) * p.vrs + mloc];
      }
      bv.u[vi] = pack2(f0, f1);
    }
    oacc = __builtin_amdgcn_wmma_f32_16x16x32_bf16(false, ap.v, false, bv.v, (short)0, oacc, false, false);
  }

#pragma unroll
  for (int r = 0; r < 8; ++r) {
    int row = q0 + r + 8 * khalf;
    if (active && row < p.T) {
      float l = lrow[r];
      float v = (l > 1e-30f) ? oacc[r] / l : 0.f;
      p.O[(long long)n * p.sOn + (long long)row * p.ors + h * 16 + mloc] = v;
    }
  }
}

// ---------------------------------------------------------------------------
// LayerNorm over d=128 (one row per block of 128 threads), optional residual/act
// ---------------------------------------------------------------------------
__global__ __launch_bounds__(128) void ln_k(const float* X, const float* RES,
                                            const float* w, const float* b,
                                            float* Y, int act) {
  int row = blockIdx.x, t = threadIdx.x;
  long long idx = (long long)row * 128 + t;
  float v = X[idx];
  if (RES) v += RES[idx];
  __shared__ float r1[4], r2[4];
  float s = v;
  s += __shfl_xor(s, 1, 32); s += __shfl_xor(s, 2, 32); s += __shfl_xor(s, 4, 32);
  s += __shfl_xor(s, 8, 32); s += __shfl_xor(s, 16, 32);
  if ((t & 31) == 0) r1[t >> 5] = s;
  __syncthreads();
  float mean = (r1[0] + r1[1] + r1[2] + r1[3]) * (1.f / 128.f);
  float d = v - mean;
  float s2 = d * d;
  s2 += __shfl_xor(s2, 1, 32); s2 += __shfl_xor(s2, 2, 32); s2 += __shfl_xor(s2, 4, 32);
  s2 += __shfl_xor(s2, 8, 32); s2 += __shfl_xor(s2, 16, 32);
  if ((t & 31) == 0) r2[t >> 5] = s2;
  __syncthreads();
  float var = (r2[0] + r2[1] + r2[2] + r2[3]) * (1.f / 128.f);
  float y = d * rsqrtf(var + 1e-5f) * w[t] + b[t];
  Y[idx] = actf(y, act);
}

// ---------------------------------------------------------------------------
// Centered moving average (k=13, replicate pad) + residual split
// ---------------------------------------------------------------------------
__global__ void movavg_k(const float* h, float* xt, float* xw) {
  long long i = (long long)blockIdx.x * blockDim.x + threadIdx.x;
  if (i >= 64LL * 512 * 128) return;
  int c = (int)(i % 128);
  int t = (int)((i / 128) % 512);
  long long nb = i / (128 * 512);
  const float* base = h + nb * 512 * 128;
  float s = 0.f;
#pragma unroll
  for (int j = -6; j <= 6; ++j) {
    int tc = t + j; tc = tc < 0 ? 0 : (tc > 511 ? 511 : tc);
    s += base[(long long)tc * 128 + c];
  }
  float m = s * (1.f / 13.f);
  xt[i] = m;
  xw[i] = h[i] - m;
}

// ---------------------------------------------------------------------------
// Spectral mask: mag vs theta * local-mean (k=64, zero pad), split cyc/vol
// ---------------------------------------------------------------------------
__global__ void spec_k(const float* Re, const float* Im, const float* theta,
                       float* cRe, float* cIm, float* vRe, float* vIm) {
  long long i = (long long)blockIdx.x * blockDim.x + threadIdx.x;
  if (i >= 64LL * 257 * 128) return;
  int c = (int)(i % 128);
  int f = (int)((i / 128) % 257);
  long long nb = i / (128 * 257);
  const float* Rb = Re + nb * 257 * 128;
  const float* Ib = Im + nb * 257 * 128;
  float re = Re[i], im = Im[i];
  float mag = sqrtf(re * re + im * im);
  float acc = 0.f;
  for (int u = f - 32; u < f + 32; ++u) {
    if (u >= 0 && u < 257) {
      float rr = Rb[(long long)u * 128 + c];
      float ii = Ib[(long long)u * 128 + c];
      acc += sqrtf(rr * rr + ii * ii);
    }
  }
  float local = acc * (1.f / 64.f);
  float m = (mag > theta[0] * local) ? 1.f : 0.f;
  cRe[i] = re * m;         cIm[i] = im * m;
  vRe[i] = re * (1.f - m); vIm[i] = im * (1.f - m);
}

// ---------------------------------------------------------------------------
// RevIN over time axis (L=512, unbiased std): one wave per (n,d)
// ---------------------------------------------------------------------------
__global__ __launch_bounds__(256) void revin_k(const float* X, float* Y,
                                               const float* w, const float* b) {
  int wave = threadIdx.x >> 5, lane = threadIdx.x & 31;
  int pair = blockIdx.x * 8 + wave;            // 8192 pairs
  int n = pair >> 7, d = pair & 127;
  const float* base = X + (long long)n * 512 * 128 + d;
  float vals[16];
  float s = 0.f;
#pragma unroll
  for (int i = 0; i < 16; ++i) {
    vals[i] = base[(long long)(lane + 32 * i) * 128];
    s += vals[i];
  }
  s += __shfl_xor(s, 1, 32); s += __shfl_xor(s, 2, 32); s += __shfl_xor(s, 4, 32);
  s += __shfl_xor(s, 8, 32); s += __shfl_xor(s, 16, 32);
  float mean = s * (1.f / 512.f);
  float q = 0.f;
#pragma unroll
  for (int i = 0; i < 16; ++i) { float dd = vals[i] - mean; q += dd * dd; }
  q += __shfl_xor(q, 1, 32); q += __shfl_xor(q, 2, 32); q += __shfl_xor(q, 4, 32);
  q += __shfl_xor(q, 8, 32); q += __shfl_xor(q, 16, 32);
  float sd = sqrtf(q * (1.f / 511.f)) + 1e-5f;
  float sc = w[d] / sd, bb = b[d];
  float* out = Y + (long long)n * 512 * 128 + d;
#pragma unroll
  for (int i = 0; i < 16; ++i)
    out[(long long)(lane + 32 * i) * 128] = (vals[i] - mean) * sc + bb;
}

// ---------------------------------------------------------------------------
// LSTM recurrence (gates precomputed into xg by GEMM); one block per sample
// ---------------------------------------------------------------------------
__global__ __launch_bounds__(512) void lstm_k(const float* xg, const float* Whh,
                                              const float* bhh, float* hs) {
  __shared__ float hsh[128], csh[128], gsh[512];
  int n = blockIdx.x, g = threadIdx.x;
  if (g < 128) { hsh[g] = 0.f; csh[g] = 0.f; }
  __syncthreads();
  const float* xgb = xg + (long long)n * 512 * 512;
  const float* wr = Whh + (long long)g * 128;
  float bb = bhh[g];
  for (int t = 0; t < 512; ++t) {
    float acc = xgb[(long long)t * 512 + g] + bb;
#pragma unroll 8
    for (int k = 0; k < 128; ++k) acc += hsh[k] * wr[k];
    gsh[g] = acc;
    __syncthreads();
    if (g < 128) {
      float i_ = 1.f / (1.f + __expf(-gsh[g]));
      float f_ = 1.f / (1.f + __expf(-gsh[128 + g]));
      float gg = tanhf(gsh[256 + g]);
      float o_ = 1.f / (1.f + __expf(-gsh[384 + g]));
      float c = f_ * csh[g] + i_ * gg;
      csh[g] = c;
      float hh = o_ * tanhf(c);
      hsh[g] = hh;
      hs[(long long)n * 512 * 128 + (long long)t * 128 + g] = hh;
    }
    __syncthreads();
  }
}

// ---------------------------------------------------------------------------
// Fourier attention scores: softmax over Dh=16 (16-lane groups), gate V
// ---------------------------------------------------------------------------
__global__ void fscore_k(const float* Qre, const float* Qim,
                         const float* Kre, const float* Kim,
                         const float* Vre, const float* Vim,
                         float* Ore, float* Oim) {
  long long tidg = (long long)blockIdx.x * 256 + threadIdx.x;
  long long gid = tidg >> 4;
  if (gid >= 512LL * 257) return;
  long long i = tidg;
  float s = (Qre[i] * Kre[i] + Qim[i] * Kim[i]) * 0.25f;
  float mt = s;
  mt = fmaxf(mt, __shfl_xor(mt, 1, 16));
  mt = fmaxf(mt, __shfl_xor(mt, 2, 16));
  mt = fmaxf(mt, __shfl_xor(mt, 4, 16));
  mt = fmaxf(mt, __shfl_xor(mt, 8, 16));
  float p = __expf(s - mt);
  float ps = p;
  ps += __shfl_xor(ps, 1, 16);
  ps += __shfl_xor(ps, 2, 16);
  ps += __shfl_xor(ps, 4, 16);
  ps += __shfl_xor(ps, 8, 16);
  float a = p / ps;
  Ore[i] = a * Vre[i];
  Oim[i] = a * Vim[i];
}

// ---------------------------------------------------------------------------
// Misc small kernels
// ---------------------------------------------------------------------------
__global__ void pool_k(const float* Z, float* zp) {
  int n = blockIdx.x, d = threadIdx.x;
  const float* base = Z + (long long)n * 512 * 128 + d;
  float s = 0.f;
  for (int t = 0; t < 512; ++t) s += base[(long long)t * 128];
  zp[n * 128 + d] = s * (1.f / 512.f);
}

__global__ void fuse_k(const float* g1, const float* a1, const float* g2,
                       const float* a2, const float* g3, const float* a3,
                       float* z) {
  long long i = (long long)blockIdx.x * blockDim.x + threadIdx.x;
  if (i >= 64LL * 512 * 128) return;
  z[i] = g1[i] * a1[i] + g2[i] * a2[i] + g3[i] * a3[i];
}

// DFT tables: rfft (257x512) and irfft (512x257) as dense matrices
__global__ void dft_k(float* Cf, float* Sf, float* ICr, float* ICi) {
  int i = blockIdx.x * 256 + threadIdx.x;
  if (i >= 257 * 512) return;
  int f = i / 512, t = i % 512;
  float ang = 6.283185307179586f * (float)((f * t) & 511) * (1.f / 512.f);
  Cf[i] = cosf(ang);
  Sf[i] = -sinf(ang);
  int tt = i / 257, ff = i % 257;
  float ang2 = 6.283185307179586f * (float)((ff * tt) & 511) * (1.f / 512.f);
  float wf = (ff == 0 || ff == 256) ? 1.f : 2.f;
  ICr[i] = wf * cosf(ang2) * (1.f / 512.f);
  ICi[i] = -wf * sinf(ang2) * (1.f / 512.f);
}

// ---------------------------------------------------------------------------
// Host helpers
// ---------------------------------------------------------------------------
static void gemm(hipStream_t st, const float* A, int lda, long long sAo, long long sAi,
                 const float* B, int ldb, long long sBo, long long sBi, int opBT,
                 const float* bias, float* C, int ldc, long long sCo, long long sCi,
                 int M, int N, int K, int outer, int inner, int act, int beta) {
  GemmP p;
  p.A = A; p.B = B; p.bias = bias; p.C = C;
  p.sAo = sAo; p.sAi = sAi; p.sBo = sBo; p.sBi = sBi; p.sCo = sCo; p.sCi = sCi;
  p.M = M; p.N = N; p.K = K; p.lda = lda; p.ldb = ldb; p.ldc = ldc;
  p.inner = inner < 1 ? 1 : inner; p.opBT = opBT; p.act = act; p.beta = beta;
  dim3 g((unsigned)((N + 63) / 64), (unsigned)((M + 127) / 128),
         (unsigned)(outer * p.inner));
  gemm_k<<<g, dim3(256), 0, st>>>(p);
}

static void flash(hipStream_t st, const float* Q, const float* K, const float* V,
                  float* O, long long sQn, int qrs, long long sKn, int krs,
                  long long sVn, int vrs, long long sOn, int ors,
                  int N, int T, int H) {
  FlashP p;
  p.Q = Q; p.K = K; p.V = V; p.O = O;
  p.sQn = sQn; p.sKn = sKn; p.sVn = sVn; p.sOn = sOn;
  p.qrs = qrs; p.krs = krs; p.vrs = vrs; p.ors = ors;
  p.N = N; p.T = T; p.H = H;
  int nQt = (T + 15) / 16;
  long long total = (long long)N * H * nQt;
  int blocks = (int)((total + 7) / 8);
  flash_k<<<blocks, 256, 0, st>>>(p);
}

static void ln(hipStream_t st, const float* X, const float* RES, const float* w,
               const float* b, float* Y, int rows, int act) {
  ln_k<<<rows, 128, 0, st>>>(X, RES, w, b, Y, act);
}

// ---------------------------------------------------------------------------
// Full forward pass
// ---------------------------------------------------------------------------
extern "C" void kernel_launch(void* const* d_in, const int* in_sizes, int n_in,
                              void* d_out, int out_size, void* d_ws, size_t ws_size,
                              hipStream_t stream) {
  (void)in_sizes; (void)n_in; (void)out_size; (void)ws_size;
  float* ws = (float*)d_ws;
  auto P = [&](int i) { return (const float*)d_in[i]; };

  const long long SZ_DFT = 257LL * 512;        // 131584
  const long long SZ_NLD = 64LL * 512 * 128;   // 4194304
  const long long SZ_SP  = 64LL * 257 * 128;   // 2105344 (== 512*257*16)

  float* Cf   = ws;
  float* Sf   = Cf + SZ_DFT;
  float* ICr  = Sf + SZ_DFT;
  float* ICi  = ICr + SZ_DFT;
  float* Hbuf = ICi + SZ_DFT;
  float* XT   = Hbuf + SZ_NLD;
  float* XW   = XT + SZ_NLD;
  float* XC   = XW + SZ_NLD;
  float* XV   = XC + SZ_NLD;
  float* HT   = XV + SZ_NLD;
  float* HC   = HT + SZ_NLD;
  float* HV   = HC + SZ_NLD;
  float* T1   = HV + SZ_NLD;                   // 64*512*512 floats
  float* T2   = T1 + 64LL * 512 * 512;
  float* T3   = T2 + SZ_NLD;
  float* T4   = T3 + SZ_NLD;
  float* T5   = T4 + SZ_NLD;

  // DFT / IDFT matrices
  dft_k<<<(int)((SZ_DFT + 255) / 256), 256, 0, stream>>>(Cf, Sf, ICr, ICi);

  // input projection: (B*M*L, 8) @ (8,128) + b
  gemm(stream, P(0), 8, 0, 0, P(1), 128, 0, 0, 0, P(2),
       Hbuf, 128, 0, 0, 32768, 128, 8, 1, 1, 0, 0);

  // trend / residual decomposition
  movavg_k<<<(int)(SZ_NLD / 256), 256, 0, stream>>>(Hbuf, XT, XW);

  // spectral split: rfft(x_w) as DFT GEMMs, adaptive mask, irfft
  float* XwRe = T1 + 0 * SZ_SP; float* XwIm = T1 + 1 * SZ_SP;
  float* XcRe = T1 + 2 * SZ_SP; float* XcIm = T1 + 3 * SZ_SP;
  float* XvRe = T1 + 4 * SZ_SP; float* XvIm = T1 + 5 * SZ_SP;
  gemm(stream, Cf, 512, 0, 0, XW, 128, 512LL * 128, 0, 0, nullptr,
       XwRe, 128, 257LL * 128, 0, 257, 128, 512, 64, 1, 0, 0);
  gemm(stream, Sf, 512, 0, 0, XW, 128, 512LL * 128, 0, 0, nullptr,
       XwIm, 128, 257LL * 128, 0, 257, 128, 512, 64, 1, 0, 0);
  spec_k<<<(int)(SZ_SP / 256), 256, 0, stream>>>(XwRe, XwIm, P(3),
                                                 XcRe, XcIm, XvRe, XvIm);
  gemm(stream, ICr, 257, 0, 0, XcRe, 128, 257LL * 128, 0, 0, nullptr,
       XC, 128, 512LL * 128, 0, 512, 128, 257, 64, 1, 0, 0);
  gemm(stream, ICi, 257, 0, 0, XcIm, 128, 257LL * 128, 0, 0, nullptr,
       XC, 128, 512LL * 128, 0, 512, 128, 257, 64, 1, 0, 1);
  gemm(stream, ICr, 257, 0, 0, XvRe, 128, 257LL * 128, 0, 0, nullptr,
       XV, 128, 512LL * 128, 0, 512, 128, 257, 64, 1, 0, 0);
  gemm(stream, ICi, 257, 0, 0, XvIm, 128, 257LL * 128, 0, 0, nullptr,
       XV, 128, 512LL * 128, 0, 512, 128, 257, 64, 1, 0, 1);

  // ---- trend encoder: 2 layers (MHA + relu FFN, post-norm) ----
  const float* cur = XT;
  for (int l = 0; l < 2; ++l) {
    int pb = 4 + l * 12;
    gemm(stream, cur, 128, 0, 0, P(pb + 0), 384, 0, 0, 0, P(pb + 1),
         T1, 384, 0, 0, 32768, 384, 128, 1, 1, 0, 0);
    flash(stream, T1, T1 + 128, T1 + 256, T2,
          512LL * 384, 384, 512LL * 384, 384, 512LL * 384, 384,
          512LL * 128, 128, 64, 512, 8);
    gemm(stream, T2, 128, 0, 0, P(pb + 2), 128, 0, 0, 0, P(pb + 3),
         T3, 128, 0, 0, 32768, 128, 128, 1, 1, 0, 0);
    ln(stream, T3, cur, P(pb + 8), P(pb + 9), HT, 32768, 0);
    gemm(stream, HT, 128, 0, 0, P(pb + 4), 256, 0, 0, 0, P(pb + 5),
         T1, 256, 0, 0, 32768, 256, 128, 1, 1, 1, 0);
    gemm(stream, T1, 256, 0, 0, P(pb + 6), 128, 0, 0, 0, P(pb + 7),
         T2, 128, 0, 0, 32768, 128, 256, 1, 1, 0, 0);
    ln(stream, T2, HT, P(pb + 10), P(pb + 11), HT, 32768, 0);
    cur = HT;
  }

  // ---- cyclic encoder: 2 layers (Fourier attention + gelu FFN) ----
  cur = XC;
  for (int l = 0; l < 2; ++l) {
    int pb = 28 + l * 16;
    gemm(stream, cur, 128, 0, 0, P(pb + 0), 128, 0, 0, 0, P(pb + 1),
         T2, 128, 0, 0, 32768, 128, 128, 1, 1, 0, 0);
    gemm(stream, cur, 128, 0, 0, P(pb + 2), 128, 0, 0, 0, P(pb + 3),
         T3, 128, 0, 0, 32768, 128, 128, 1, 1, 0, 0);
    gemm(stream, cur, 128, 0, 0, P(pb + 4), 128, 0, 0, 0, P(pb + 5),
         T4, 128, 0, 0, 32768, 128, 128, 1, 1, 0, 0);
    float* QFre = T1 + 0 * SZ_SP; float* QFim = T1 + 1 * SZ_SP;
    float* KFre = T1 + 2 * SZ_SP; float* KFim = T1 + 3 * SZ_SP;
    float* VFre = T1 + 4 * SZ_SP; float* VFim = T1 + 5 * SZ_SP;
    const long long sBn = 512LL * 128, sBh = 16;
    const long long sCn = 8LL * 257 * 16, sCh = 257 * 16;
    gemm(stream, Cf, 512, 0, 0, T2, 128, sBn, sBh, 0, nullptr, QFre, 16, sCn, sCh, 257, 16, 512, 64, 8, 0, 0);
    gemm(stream, Sf, 512, 0, 0, T2, 128, sBn, sBh, 0, nullptr, QFim, 16, sCn, sCh, 257, 16, 512, 64, 8, 0, 0);
    gemm(stream, Cf, 512, 0, 0, T3, 128, sBn, sBh, 0, nullptr, KFre, 16, sCn, sCh, 257, 16, 512, 64, 8, 0, 0);
    gemm(stream, Sf, 512, 0, 0, T3, 128, sBn, sBh, 0, nullptr, KFim, 16, sCn, sCh, 257, 16, 512, 64, 8, 0, 0);
    gemm(stream, Cf, 512, 0, 0, T4, 128, sBn, sBh, 0, nullptr, VFre, 16, sCn, sCh, 257, 16, 512, 64, 8, 0, 0);
    gemm(stream, Sf, 512, 0, 0, T4, 128, sBn, sBh, 0, nullptr, VFim, 16, sCn, sCh, 257, 16, 512, 64, 8, 0, 0);
    float* ORe = T2; float* OIm = T3;
    fscore_k<<<(int)((512LL * 257 * 16) / 256), 256, 0, stream>>>(
        QFre, QFim, KFre, KFim, VFre, VFim, ORe, OIm);
    gemm(stream, ICr, 257, 0, 0, ORe, 16, sCn, sCh, 0, nullptr,
         T4, 128, 512LL * 128, 16, 512, 16, 257, 64, 8, 0, 0);
    gemm(stream, ICi, 257, 0, 0, OIm, 16, sCn, sCh, 0, nullptr,
         T4, 128, 512LL * 128, 16, 512, 16, 257, 64, 8, 0, 1);
    gemm(stream, T4, 128, 0, 0, P(pb + 6), 128, 0, 0, 0, P(pb + 7),
         T5, 128, 0, 0, 32768, 128, 128, 1, 1, 0, 0);
    ln(stream, T5, cur, P(pb + 12), P(pb + 13), HC, 32768, 0);
    gemm(stream, HC, 128, 0, 0, P(pb + 8), 256, 0, 0, 0, P(pb + 9),
         T1, 256, 0, 0, 32768, 256, 128, 1, 1, 2, 0);
    gemm(stream, T1, 256, 0, 0, P(pb + 10), 128, 0, 0, 0, P(pb + 11),
         T2, 128, 0, 0, 32768, 128, 256, 1, 1, 0, 0);
    ln(stream, T2, HC, P(pb + 14), P(pb + 15), HC, 32768, 0);
    cur = HC;
  }

  // ---- volatility encoder: RevIN -> MLP -> LSTM -> RevIN -> MLP ----
  revin_k<<<1024, 256, 0, stream>>>(XV, HV, P(60), P(61));
  gemm(stream, HV, 128, 0, 0, P(62), 256, 0, 0, 0, P(63),
       T1, 256, 0, 0, 32768, 256, 128, 1, 1, 1, 0);
  gemm(stream, T1, 256, 0, 0, P(64), 128, 0, 0, 0, P(65),
       T2, 128, 0, 0, 32768, 128, 256, 1, 1, 0, 0);
  gemm(stream, T2, 128, 0, 0, P(66), 128, 0, 0, 1, P(68),   // xg = x @ Wih^T + bih
       T1, 512, 0, 0, 32768, 512, 128, 1, 1, 0, 0);
  lstm_k<<<64, 512, 0, stream>>>(T1, P(67), P(69), T3);
  revin_k<<<1024, 256, 0, stream>>>(T3, T2, P(70), P(71));
  gemm(stream, T2, 128, 0, 0, P(72), 256, 0, 0, 0, P(73),
       T1, 256, 0, 0, 32768, 256, 128, 1, 1, 1, 0);
  gemm(stream, T1, 256, 0, 0, P(74), 128, 0, 0, 0, P(75),
       HV, 128, 0, 0, 32768, 128, 256, 1, 1, 0, 0);

  // ---- AMCI gated fusion ----
  gemm(stream, HT, 128, 0, 0, P(76), 128, 0, 0, 0, P(77),
       T1, 128, 0, 0, 32768, 128, 128, 1, 1, 3, 0);
  gemm(stream, HC, 128, 0, 0, P(78), 128, 0, 0, 0, P(79),
       T1 + SZ_NLD, 128, 0, 0, 32768, 128, 128, 1, 1, 3, 0);
  gemm(stream, HV, 128, 0, 0, P(80), 128, 0, 0, 0, P(81),
       T1 + 2 * SZ_NLD, 128, 0, 0, 32768, 128, 128, 1, 1, 3, 0);
  fuse_k<<<(int)(SZ_NLD / 256), 256, 0, stream>>>(T1, HT, T1 + SZ_NLD, HC,
                                                  T1 + 2 * SZ_NLD, HV, T2);

  // ---- pool + temporal projection (gelu(LN(zp @ W + b))) ----
  pool_k<<<64, 128, 0, stream>>>(T2, T3);
  gemm(stream, T3, 128, 0, 0, P(82), 128, 0, 0, 0, P(83),
       T4, 128, 0, 0, 64, 128, 128, 1, 1, 0, 0);
  ln(stream, T4, nullptr, P(84), P(85), T5, 64, 2);

  // ---- ISCA cross-ETF attention (B=8 batches of T=8 tokens) + classifier ----
  {
    int pb = 86;
    gemm(stream, T5, 128, 0, 0, P(pb + 0), 384, 0, 0, 0, P(pb + 1),
         T1, 384, 0, 0, 64, 384, 128, 1, 1, 0, 0);
    flash(stream, T1, T1 + 128, T1 + 256, T2,
          8LL * 384, 384, 8LL * 384, 384, 8LL * 384, 384,
          8LL * 128, 128, 8, 8, 8);
    gemm(stream, T2, 128, 0, 0, P(pb + 2), 128, 0, 0, 0, P(pb + 3),
         T3, 128, 0, 0, 64, 128, 128, 1, 1, 0, 0);
    ln(stream, T3, T5, P(pb + 8), P(pb + 9), T4, 64, 0);
    gemm(stream, T4, 128, 0, 0, P(pb + 4), 256, 0, 0, 0, P(pb + 5),
         T1, 256, 0, 0, 64, 256, 128, 1, 1, 2, 0);
    gemm(stream, T1, 256, 0, 0, P(pb + 6), 128, 0, 0, 0, P(pb + 7),
         T2, 128, 0, 0, 64, 128, 256, 1, 1, 0, 0);
    ln(stream, T2, T4, P(pb + 10), P(pb + 11), T3, 64, 0);
    gemm(stream, T3, 128, 0, 0, P(98), 2, 0, 0, 0, P(99),
         (float*)d_out, 2, 0, 0, 64, 2, 128, 1, 1, 0, 0);
  }
}
